// OPTAttention_72524817760265
// MI455X (gfx1250) — matrix operand order repaired and checked by reading, not compiled, not run on hardware
//
#include <hip/hip_runtime.h>

// ---------------------------------------------------------------------------
// OPT attention block for MI455X (gfx1250, wave32, WMMA + TDM).
// Pipeline: f32->bf16 convert -> QKV GEMMs (WMMA bf16) -> flash attention
// (TDM-staged K/V tiles in LDS, WMMA bf16, LDS bounce for P) -> output GEMM.
// ---------------------------------------------------------------------------

#define DEVI __device__ __forceinline__

typedef __bf16 bf16_t;
typedef bf16_t v16bf __attribute__((ext_vector_type(16)));
typedef float  v8f   __attribute__((ext_vector_type(8)));
typedef unsigned int u32x4 __attribute__((ext_vector_type(4)));
typedef int          i32x8 __attribute__((ext_vector_type(8)));
typedef int          i32x4 __attribute__((ext_vector_type(4)));

struct alignas(16) U128 { unsigned int w[4]; };
union Frag { U128 h[2]; v16bf v; };

static constexpr int   Bb   = 2;
static constexpr int   Tt   = 2048;
static constexpr int   Dd   = 2048;
static constexpr int   Hh   = 32;
static constexpr int   HDh  = 64;
static constexpr float SCALE = 0.125f;           // 64^-0.5
static constexpr float NEGF  = -3.402823466e38f; // finfo(f32).min

DEVI unsigned short f2bf(float f) {
  unsigned int u = __float_as_uint(f);
  u += 0x7FFFu + ((u >> 16) & 1u);   // round-to-nearest-even
  return (unsigned short)(u >> 16);
}

// A-matrix fragment (16x32, M x K), row-major source, k contiguous.
DEVI v16bf load_a_frag(const unsigned short* base, int ld, int row0, int k0) {
  const int lane = threadIdx.x & 31;
  const int half = lane >> 4, m = lane & 15;
  const unsigned short* p = base + (size_t)(row0 + m) * ld + (k0 + half * 16);
  Frag f;
  f.h[0] = *(const U128*)(p);
  f.h[1] = *(const U128*)(p + 8);
  return f.v;
}

// B-matrix fragment (32x16, K x N) where B[k][n] = Src[col0+n][k] (row-major NxK).
DEVI v16bf load_b_frag(const unsigned short* base, int ld, int col0, int k0) {
  const int lane = threadIdx.x & 31;
  const int half = lane >> 4, n = lane & 15;
  const unsigned short* p = base + (size_t)(col0 + n) * ld + (k0 + half * 16);
  Frag f;
  f.h[0] = *(const U128*)(p);
  f.h[1] = *(const U128*)(p + 8);
  return f.v;
}

DEVI v8f wmma_bf16(v16bf a, v16bf b, v8f c) {
  return __builtin_amdgcn_wmma_f32_16x16x32_bf16(
      false, a, false, b, (short)0, c, false, false);
}

// ---------------------------------------------------------------------------
// Tensor Data Mover: stage a 2D tile (th rows of tw bf16 elements, row stride
// `stride` elements) from global memory into LDS at byte offset lds_off.
// D# built per CDNA5 ISA (tail groups zero => <=2D tile).
// 6-arg builtin form (clang-23 / amdgpu-toolchain).
// ---------------------------------------------------------------------------
DEVI void tdm_load_tile(unsigned lds_off, const unsigned short* gptr,
                        unsigned tw, unsigned th, unsigned stride) {
  unsigned long long ga = (unsigned long long)(uintptr_t)gptr;
  u32x4 g0;
  g0[0] = 1u;                                     // count=1, user-mode D#
  g0[1] = lds_off;                                // lds_addr   (bits 63:32)
  g0[2] = (unsigned)(ga & 0xFFFFFFFFu);           // global_addr[31:0]  (95:64)
  g0[3] = (unsigned)((ga >> 32) & 0x01FFFFFFu)    // global_addr[56:32] (120:96)
        | (2u << 30);                             // type=2 "image"     (127:126)
  i32x8 g1;
  g1[0] = 0x10000;                                // wg_mask=0, data_size=1 (2B)
  g1[1] = (int)((tw & 0xFFFFu) << 16);            // tensor_dim0[15:0]  (63:48)
  g1[2] = (int)(((tw >> 16) & 0xFFFFu)            // tensor_dim0[31:16] (79:64)
        | ((th & 0xFFFFu) << 16));                // tensor_dim1[15:0]  (95:80)
  g1[3] = (int)(((th >> 16) & 0xFFFFu)            // tensor_dim1[31:16] (111:96)
        | ((tw & 0xFFFFu) << 16));                // tile_dim0          (127:112)
  g1[4] = (int)(th & 0xFFFFu);                    // tile_dim1; tile_dim2=0
  g1[5] = (int)stride;                            // tensor_dim0_stride[31:0]
  g1[6] = 0;                                      // stride0 hi, stride1 lo
  g1[7] = 0;                                      // stride1 hi
  const i32x4 z4 = {0, 0, 0, 0};                  // unused higher-dim groups
  const i32x8 z8 = {0, 0, 0, 0, 0, 0, 0, 0};
  __builtin_amdgcn_tensor_load_to_lds(g0, g1, z4, z4, z8, 0);
}

// ---------------------------------------------------------------------------
// f32 -> bf16 elementwise conversion (vectorized x4).
// ---------------------------------------------------------------------------
__global__ void cvt_f32_bf16(const float* __restrict__ src,
                             unsigned short* __restrict__ dst, int n4) {
  int i = blockIdx.x * blockDim.x + threadIdx.x;
  if (i >= n4) return;
  float4 v = ((const float4*)src)[i];
  unsigned int lo = (unsigned int)f2bf(v.x) | ((unsigned int)f2bf(v.y) << 16);
  unsigned int hi = (unsigned int)f2bf(v.z) | ((unsigned int)f2bf(v.w) << 16);
  ((uint2*)dst)[i] = make_uint2(lo, hi);
}

// ---------------------------------------------------------------------------
// GEMM: out[m,n] = sum_k A[m,k] * W[n,k] + bias[n]; one wave = 64x64 tile.
// mode 0: Q->(B,H,T,HD)*SCALE  1: K->(B,H,T,HD)  2: V->(B,H,HD,T)  3: f32 out
// ---------------------------------------------------------------------------
__global__ void __launch_bounds__(32)
gemm_qkvo(const unsigned short* __restrict__ A, const unsigned short* __restrict__ W,
          const float* __restrict__ bias, unsigned short* __restrict__ outb,
          float* __restrict__ outf, int mode) {
  const int K = Dd;
  const int aRow = blockIdx.y * 64;
  const int bCol = blockIdx.x * 64;

  v8f acc[4][4];
  const v8f vzero = {0.f, 0.f, 0.f, 0.f, 0.f, 0.f, 0.f, 0.f};
#pragma unroll
  for (int i = 0; i < 4; ++i)
#pragma unroll
    for (int j = 0; j < 4; ++j) acc[i][j] = vzero;

  for (int k0 = 0; k0 < K; k0 += 32) {
    v16bf af[4], bfr[4];
#pragma unroll
    for (int i = 0; i < 4; ++i) af[i] = load_a_frag(A, K, aRow + 16 * i, k0);
#pragma unroll
    for (int j = 0; j < 4; ++j) bfr[j] = load_b_frag(W, K, bCol + 16 * j, k0);
#pragma unroll
    for (int i = 0; i < 4; ++i)
#pragma unroll
      for (int j = 0; j < 4; ++j) acc[i][j] = wmma_bf16(af[i], bfr[j], acc[i][j]);
  }

  const int lane = threadIdx.x & 31, half = lane >> 4, n15 = lane & 15;
#pragma unroll
  for (int i = 0; i < 4; ++i)
#pragma unroll
    for (int j = 0; j < 4; ++j)
#pragma unroll
      for (int r = 0; r < 8; ++r) {
        int Mr = aRow + 16 * i + r + 8 * half;  // row in [0, B*T)
        int Nc = bCol + 16 * j + n15;           // col in [0, D)
        float val = acc[i][j][r] + bias[Nc];
        if (mode == 3) {
          outf[(size_t)Mr * Dd + Nc] = val;
        } else {
          int b = Mr >> 11, t = Mr & (Tt - 1);
          int h = Nc >> 6, hd = Nc & (HDh - 1);
          if (mode == 0)
            outb[((size_t)(b * Hh + h) * Tt + t) * HDh + hd] = f2bf(val * SCALE);
          else if (mode == 1)
            outb[((size_t)(b * Hh + h) * Tt + t) * HDh + hd] = f2bf(val);
          else
            outb[((size_t)(b * Hh + h) * HDh + hd) * Tt + t] = f2bf(val);
        }
      }
}

// ---------------------------------------------------------------------------
// Flash attention: one wave per (b, h, 64-row q tile).
// K/V tiles are DMA'd into double-buffered LDS by the TDM (overlapping the
// WMMA work on the previous tile); P is bounced through LDS for layout change.
// ---------------------------------------------------------------------------
__global__ void __launch_bounds__(32)
attn_fa(const unsigned short* __restrict__ qb, const unsigned short* __restrict__ kbuf,
        const unsigned short* __restrict__ vtb, const float* __restrict__ mask,
        unsigned short* __restrict__ ctxb) {
  __shared__ unsigned short Kst[2][64 * 64];   // staged K tile: [key][hd]
  __shared__ unsigned short Vst[2][64 * 64];   // staged V tile: [hd][key]
  __shared__ unsigned short Plds[64 * 80];     // P bounce, 160B row stride

  const int bh = blockIdx.y;
  const int b = bh >> 5, h = bh & (Hh - 1);
  const int qbase = blockIdx.x * 64;
  const unsigned short* Q  = qb   + (size_t)bh * Tt * HDh;
  const unsigned short* Kp = kbuf + (size_t)bh * Tt * HDh;
  const unsigned short* Vt = vtb  + (size_t)bh * HDh * Tt;
  const float* Mk = mask + (size_t)b * Tt * Tt;

  const int lane = threadIdx.x & 31, half = lane >> 4, n15 = lane & 15;
  const unsigned kst0 = (unsigned)(uintptr_t)&Kst[0][0];
  const unsigned kst1 = (unsigned)(uintptr_t)&Kst[1][0];
  const unsigned vst0 = (unsigned)(uintptr_t)&Vst[0][0];
  const unsigned vst1 = (unsigned)(uintptr_t)&Vst[1][0];

  // Preload Q fragments for this q tile (A-layout, lda = HD).
  v16bf qf[4][2];
#pragma unroll
  for (int mi = 0; mi < 4; ++mi) {
    qf[mi][0] = load_a_frag(Q, HDh, qbase + 16 * mi, 0);
    qf[mi][1] = load_a_frag(Q, HDh, qbase + 16 * mi, 32);
  }

  v8f ctx[4][4];
  const v8f vzero = {0.f, 0.f, 0.f, 0.f, 0.f, 0.f, 0.f, 0.f};
  float mrow[4][8], lrow[4][8];
#pragma unroll
  for (int mi = 0; mi < 4; ++mi) {
#pragma unroll
    for (int j = 0; j < 4; ++j) ctx[mi][j] = vzero;
#pragma unroll
    for (int r = 0; r < 8; ++r) { mrow[mi][r] = NEGF; lrow[mi][r] = 0.f; }
  }

  // Kick off DMA of tile 0 into buffer 0.
  // K tile: 64 contiguous rows of 64 bf16 => one 4096-element line.
  // V tile: 64 rows (hd) x 64 cols (key), row stride T.
  tdm_load_tile(kst0, Kp, 4096u, 1u, 4096u);
  tdm_load_tile(vst0, Vt, 64u, 64u, (unsigned)Tt);

  const int sEnd = qbase + 64;                 // causal: skip tiles past diagonal
  for (int s0 = 0; s0 < sEnd; s0 += 64) {
    const int bi = (s0 >> 6) & 1;
    const unsigned short* Kt = &Kst[bi][0];
    const unsigned short* Vs = &Vst[bi][0];
    if (s0 + 64 < sEnd) {                      // prefetch next tile into other buf
      int s1 = s0 + 64;
      tdm_load_tile(bi ? kst0 : kst1, Kp + (size_t)s1 * HDh, 4096u, 1u, 4096u);
      tdm_load_tile(bi ? vst0 : vst1, Vt + s1, 64u, 64u, (unsigned)Tt);
      __builtin_amdgcn_s_wait_tensorcnt(2);    // current tile's 2 DMAs complete
    } else {
      __builtin_amdgcn_s_wait_tensorcnt(0);
    }

    // ---- S = Q K^T  (K B-frags from LDS) ----
    v8f sc[4][4];
#pragma unroll
    for (int ni = 0; ni < 4; ++ni) {
      Frag f0, f1;
      {
        const unsigned short* p = &Kt[(16 * ni + n15) * 64 + half * 16];
        f0.h[0] = *(const U128*)(p);
        f0.h[1] = *(const U128*)(p + 8);
        p += 32;
        f1.h[0] = *(const U128*)(p);
        f1.h[1] = *(const U128*)(p + 8);
      }
#pragma unroll
      for (int mi = 0; mi < 4; ++mi) {
        v8f z = wmma_bf16(qf[mi][0], f0.v, vzero);
        sc[mi][ni] = wmma_bf16(qf[mi][1], f1.v, z);
      }
    }
    // ---- + mask, clamp to NEG ----
#pragma unroll
    for (int mi = 0; mi < 4; ++mi)
#pragma unroll
      for (int r = 0; r < 8; ++r) {
        int qr = qbase + 16 * mi + r + 8 * half;
#pragma unroll
        for (int ni = 0; ni < 4; ++ni) {
          int kc = s0 + 16 * ni + n15;
          float v = sc[mi][ni][r] + Mk[(size_t)qr * Tt + kc];
          sc[mi][ni][r] = fmaxf(v, NEGF);
        }
      }
    // ---- online softmax (rows spread across 16 lanes of each half) ----
#pragma unroll
    for (int mi = 0; mi < 4; ++mi)
#pragma unroll
      for (int r = 0; r < 8; ++r) {
        float vmax = fmaxf(fmaxf(sc[mi][0][r], sc[mi][1][r]),
                           fmaxf(sc[mi][2][r], sc[mi][3][r]));
#pragma unroll
        for (int xm = 1; xm < 16; xm <<= 1) vmax = fmaxf(vmax, __shfl_xor(vmax, xm, 32));
        float mnew = fmaxf(mrow[mi][r], vmax);
        float corr = __expf(mrow[mi][r] - mnew);
        mrow[mi][r] = mnew;
        float psum = 0.f;
#pragma unroll
        for (int ni = 0; ni < 4; ++ni) {
          float p = __expf(sc[mi][ni][r] - mnew);
          sc[mi][ni][r] = p;
          psum += p;
        }
#pragma unroll
        for (int xm = 1; xm < 16; xm <<= 1) psum += __shfl_xor(psum, xm, 32);
        lrow[mi][r] = lrow[mi][r] * corr + psum;
#pragma unroll
        for (int j = 0; j < 4; ++j) ctx[mi][j][r] *= corr;
      }
    // ---- P (C-layout) -> LDS (row-major bf16) ----
#pragma unroll
    for (int mi = 0; mi < 4; ++mi)
#pragma unroll
      for (int ni = 0; ni < 4; ++ni)
#pragma unroll
        for (int r = 0; r < 8; ++r) {
          int row = 16 * mi + r + 8 * half, col = 16 * ni + n15;
          Plds[row * 80 + col] = f2bf(sc[mi][ni][r]);
        }
    __syncthreads();
    // ---- ctx += P V  (P A-frags from LDS, V B-frags from staged LDS tile) ----
#pragma unroll
    for (int kc = 0; kc < 2; ++kc) {
      v16bf pa[4];
#pragma unroll
      for (int mi = 0; mi < 4; ++mi) {
        const unsigned short* p = &Plds[(16 * mi + n15) * 80 + kc * 32 + half * 16];
        Frag f;
        f.h[0] = *(const U128*)(p);
        f.h[1] = *(const U128*)(p + 8);
        pa[mi] = f.v;
      }
#pragma unroll
      for (int ni = 0; ni < 4; ++ni) {
        Frag vf;
        const unsigned short* p = &Vs[(16 * ni + n15) * 64 + kc * 32 + half * 16];
        vf.h[0] = *(const U128*)(p);
        vf.h[1] = *(const U128*)(p + 8);
#pragma unroll
        for (int mi = 0; mi < 4; ++mi) ctx[mi][ni] = wmma_bf16(pa[mi], vf.v, ctx[mi][ni]);
      }
    }
    __syncthreads();
  }

  // ---- normalize + store ctx (B,T,D) bf16 ----
#pragma unroll
  for (int mi = 0; mi < 4; ++mi)
#pragma unroll
    for (int r = 0; r < 8; ++r) {
      float inv = 1.f / lrow[mi][r];
      int qr = qbase + 16 * mi + r + 8 * half;
#pragma unroll
      for (int j = 0; j < 4; ++j) {
        int hd = 16 * j + n15;
        ctxb[((size_t)b * Tt + qr) * Dd + h * HDh + hd] = f2bf(ctx[mi][j][r] * inv);
      }
    }
}

// ---------------------------------------------------------------------------
// Launch: converts -> QKV GEMMs -> attention -> out projection.
// Workspace layout (bytes): xb@0(16M) wq@16M wk@24M wv@32M wo@40M
//                           q@48M k@64M vt@80M ctx@96M  (total 112 MiB)
// ---------------------------------------------------------------------------
extern "C" void kernel_launch(void* const* d_in, const int* in_sizes, int n_in,
                              void* d_out, int out_size, void* d_ws, size_t ws_size,
                              hipStream_t stream) {
  const float* x    = (const float*)d_in[0];
  const float* mask = (const float*)d_in[1];
  const float* Wq   = (const float*)d_in[2];
  const float* bq   = (const float*)d_in[3];
  const float* Wk   = (const float*)d_in[4];
  const float* bk   = (const float*)d_in[5];
  const float* Wv   = (const float*)d_in[6];
  const float* bv   = (const float*)d_in[7];
  const float* Wo   = (const float*)d_in[8];
  const float* bo   = (const float*)d_in[9];
  float* out = (float*)d_out;

  char* ws = (char*)d_ws;
  const size_t MB = 1024 * 1024;
  unsigned short* xb  = (unsigned short*)(ws + 0 * MB);
  unsigned short* wqb = (unsigned short*)(ws + 16 * MB);
  unsigned short* wkb = (unsigned short*)(ws + 24 * MB);
  unsigned short* wvb = (unsigned short*)(ws + 32 * MB);
  unsigned short* wob = (unsigned short*)(ws + 40 * MB);
  unsigned short* qbf = (unsigned short*)(ws + 48 * MB);
  unsigned short* kbf = (unsigned short*)(ws + 64 * MB);
  unsigned short* vtb = (unsigned short*)(ws + 80 * MB);
  unsigned short* ctb = (unsigned short*)(ws + 96 * MB);

  // f32 -> bf16 conversions
  cvt_f32_bf16<<<(Bb * Tt * Dd / 4 + 255) / 256, 256, 0, stream>>>(x, xb, Bb * Tt * Dd / 4);
  cvt_f32_bf16<<<(Dd * Dd / 4 + 255) / 256, 256, 0, stream>>>(Wq, wqb, Dd * Dd / 4);
  cvt_f32_bf16<<<(Dd * Dd / 4 + 255) / 256, 256, 0, stream>>>(Wk, wkb, Dd * Dd / 4);
  cvt_f32_bf16<<<(Dd * Dd / 4 + 255) / 256, 256, 0, stream>>>(Wv, wvb, Dd * Dd / 4);
  cvt_f32_bf16<<<(Dd * Dd / 4 + 255) / 256, 256, 0, stream>>>(Wo, wob, Dd * Dd / 4);

  // QKV projections (one 64x64 tile per wave)
  dim3 gg(Dd / 64, (Bb * Tt) / 64);  // (32, 64)
  gemm_qkvo<<<gg, 32, 0, stream>>>(xb, wqb, bq, qbf, nullptr, 0);
  gemm_qkvo<<<gg, 32, 0, stream>>>(xb, wkb, bk, kbf, nullptr, 1);
  gemm_qkvo<<<gg, 32, 0, stream>>>(xb, wvb, bv, vtb, nullptr, 2);

  // Flash attention per (b,h,q-tile), TDM-staged K/V
  attn_fa<<<dim3(Tt / 64, Bb * Hh), 32, 0, stream>>>(qbf, kbf, vtb, mask, ctb);

  // Output projection -> f32
  gemm_qkvo<<<gg, 32, 0, stream>>>(ctb, wob, bo, nullptr, out, 3);
}
